// QuantizedRowParallel_44805098832188
// MI455X (gfx1250) — matrix operand order repaired and checked by reading, not compiled, not run on hardware
//
#include <hip/hip_runtime.h>
#include <hip/hip_bf16.h>
#include <stdint.h>

// CDNA5 (gfx1250) wave32 WMMA types
typedef __attribute__((ext_vector_type(16))) __bf16 v16bf;
typedef __attribute__((ext_vector_type(8)))  float  v8f;

#define BM 128
#define BN 128
#define BK 64
#define LDS_STRIDE 72   // bf16 elements per row: 64 + 8 pad (16B-aligned rows, conflict-free frag loads)

__device__ __forceinline__ uint32_t pack2_bf16_rne(float a, float b) {
  // round-to-nearest-even fp32 -> bf16, packed (b in high half)
  uint32_t ua = __float_as_uint(a);
  uint32_t ub = __float_as_uint(b);
  ua += 0x7FFFu + ((ua >> 16) & 1u);
  ub += 0x7FFFu + ((ub >> 16) & 1u);
  return (ua >> 16) | (ub & 0xFFFF0000u);
}

__device__ __forceinline__ uint32_t pack2_bf16_exact(float a, float b) {
  // fp32 -> bf16 by truncation; exact when inputs are small integers (|v| <= 127)
  return (__float_as_uint(a) >> 16) | (__float_as_uint(b) & 0xFFFF0000u);
}

union Frag16 {
  uint4  q[2];
  v16bf  v;
};

__global__ __launch_bounds__(256)
void qgemm_bf16_wmma(const float* __restrict__ X,      // [M, K] fp32 (row-major)
                     const int8_t* __restrict__ W,     // [N, K] int8 (row-major)
                     const float* __restrict__ scale,  // [1]
                     const float* __restrict__ bias,   // [N]
                     float* __restrict__ out,          // [M, N]
                     int M, int N, int K) {
  __shared__ __align__(16) uint16_t Asl[BM * LDS_STRIDE]; // bf16 bits, A tile [m][k]
  __shared__ __align__(16) uint16_t Bsl[BN * LDS_STRIDE]; // bf16 bits, B tile [n][k]

  const int tid  = threadIdx.x;
  const int lane = tid & 31;
  const int wave = tid >> 5;
  const int wm   = wave & 1;   // wave M index: 0..1  (64 rows each)
  const int wn   = wave >> 1;  // wave N index: 0..3  (32 cols each)

  // blockIdx.x sweeps N (fast) so the 4MB A panel stays hot in L2; the full
  // 64MB int8 weight fits in the 192MB L2 and is re-served from L2 across
  // M-rows.  HBM traffic ~= read A once + read W once + write out once.
  const int blockN = blockIdx.x * BN;
  const int blockM = blockIdx.y * BM;

  // Per-thread global staging assignment:
  // A tile: 128 rows x 64 k fp32. 2 threads per row, 32 floats (8 float4) each.
  const int aRow  = tid >> 1;            // 0..127
  const int aCol  = (tid & 1) * 32;      // 0 or 32 (floats)
  // B tile: 128 n-rows x 64 k int8. 2 threads per row, 32 bytes (2 int4) each.
  const int bRow  = tid >> 1;            // 0..127
  const int bCol  = (tid & 1) * 32;      // 0 or 32 (bytes)

  const float*  aGlobBase = X + (size_t)(blockM + aRow) * (size_t)K + aCol;
  const int8_t* bGlobBase = W + (size_t)(blockN + bRow) * (size_t)K + bCol;

  v8f acc[4][2];
#pragma unroll
  for (int i = 0; i < 4; ++i)
#pragma unroll
    for (int j = 0; j < 2; ++j)
      acc[i][j] = (v8f)(0.0f);

  for (int k0 = 0; k0 < K; k0 += BK) {
    // ---- issue global loads into registers (overlaps previous compute) ----
    float4 aReg[8];
#pragma unroll
    for (int f = 0; f < 8; ++f)
      aReg[f] = *(const float4*)(aGlobBase + k0 + f * 4);

    int4 wReg[2];
#pragma unroll
    for (int f = 0; f < 2; ++f)
      wReg[f] = *(const int4*)(bGlobBase + k0 + f * 16);

    // prefetch next K tile while this one is consumed (global_prefetch_b8)
    if (k0 + BK < K) {
      __builtin_prefetch(aGlobBase + k0 + BK, 0, 3);
      __builtin_prefetch(bGlobBase + k0 + BK, 0, 3);
    }

    __syncthreads();  // previous iteration's compute done reading LDS

    // ---- A: fp32 -> packed bf16 (RNE) -> LDS ----
    {
      uint32_t* dst = (uint32_t*)&Asl[aRow * LDS_STRIDE + aCol];
#pragma unroll
      for (int f = 0; f < 8; ++f) {
        dst[2 * f + 0] = pack2_bf16_rne(aReg[f].x, aReg[f].y);
        dst[2 * f + 1] = pack2_bf16_rne(aReg[f].z, aReg[f].w);
      }
    }
    // ---- B: int8 -> bf16 (exact, truncating pack) -> LDS ----
    {
      uint32_t* dst = (uint32_t*)&Bsl[bRow * LDS_STRIDE + bCol];
#pragma unroll
      for (int f = 0; f < 2; ++f) {
        const int v[4] = { wReg[f].x, wReg[f].y, wReg[f].z, wReg[f].w };
#pragma unroll
        for (int d = 0; d < 4; ++d) {
          const int b0 = (v[d] << 24) >> 24;
          const int b1 = (v[d] << 16) >> 24;
          const int b2 = (v[d] <<  8) >> 24;
          const int b3 =  v[d]        >> 24;
          dst[f * 8 + 2 * d + 0] = pack2_bf16_exact((float)b0, (float)b1);
          dst[f * 8 + 2 * d + 1] = pack2_bf16_exact((float)b2, (float)b3);
        }
      }
    }

    __syncthreads();  // LDS tile ready

    // ---- compute: 2 K-substeps of 32 ----
#pragma unroll
    for (int ks = 0; ks < BK; ks += 32) {
      // B fragments (2 N-tiles): lanes 0-15 hold K=ks..ks+15, lanes 16-31 K=ks+16..ks+31
      Frag16 bf[2];
      {
        const int n  = wn * 32 + (lane & 15);
        const int kb = ks + ((lane < 16) ? 0 : 16);
        const uint16_t* p0 = &Bsl[(n +  0) * LDS_STRIDE + kb];
        const uint16_t* p1 = &Bsl[(n + 16) * LDS_STRIDE + kb];
        bf[0].q[0] = *(const uint4*)(p0);
        bf[0].q[1] = *(const uint4*)(p0 + 8);
        bf[1].q[0] = *(const uint4*)(p1);
        bf[1].q[1] = *(const uint4*)(p1 + 8);
      }
      // A fragments (4 M-tiles): lanes 0-15 K=ks+0.., lanes 16-31 K=ks+16..
      Frag16 af[4];
      {
        const int kh = ks + ((lane < 16) ? 0 : 16);
#pragma unroll
        for (int i = 0; i < 4; ++i) {
          const int m = wm * 64 + i * 16 + (lane & 15);
          const uint16_t* p = &Asl[m * LDS_STRIDE + kh];
          af[i].q[0] = *(const uint4*)(p);        // K = kh .. kh+7
          af[i].q[1] = *(const uint4*)(p + 8);    // K = kh+8 .. kh+15
        }
      }
#pragma unroll
      for (int i = 0; i < 4; ++i)
#pragma unroll
        for (int j = 0; j < 2; ++j)
          acc[i][j] = __builtin_amdgcn_wmma_f32_16x16x32_bf16(
              /*neg_a=*/false, af[i].v,
              /*neg_b=*/false, bf[j].v,
              /*c_mod=*/(short)0, acc[i][j],
              /*reuse_a=*/false, /*reuse_b=*/false);
    }
  }

  // ---- epilogue: out = acc * scale + bias ----
  const float s = scale[0];
#pragma unroll
  for (int j = 0; j < 2; ++j) {
    const int n  = blockN + wn * 32 + j * 16 + (lane & 15);
    const float bv = bias[n];
#pragma unroll
    for (int i = 0; i < 4; ++i) {
      const int mtile = blockM + wm * 64 + i * 16 + ((lane < 16) ? 0 : 8);
#pragma unroll
      for (int r = 0; r < 8; ++r) {
        const int m = mtile + r;
        out[(size_t)m * (size_t)N + n] = fmaf(acc[i][j][r], s, bv);
      }
    }
  }
}

extern "C" void kernel_launch(void* const* d_in, const int* in_sizes, int n_in,
                              void* d_out, int out_size, void* d_ws, size_t ws_size,
                              hipStream_t stream) {
  const float*  x     = (const float*)d_in[0];
  const int8_t* w     = (const int8_t*)d_in[1];
  const float*  scale = (const float*)d_in[2];
  const float*  bias  = (const float*)d_in[3];
  float* out = (float*)d_out;

  const int D_out = in_sizes[3];                       // 8192
  const int D_in  = in_sizes[1] / D_out;               // 8192
  const int M     = in_sizes[0] / D_in;                // 4*2048 = 8192

  dim3 grid(D_out / BN, M / BM);   // x = N (fast) for L2 reuse of the A panel
  qgemm_bf16_wmma<<<grid, 256, 0, stream>>>(x, w, scale, bias, out, M, D_out, D_in);
}
